// Block_67267777790222
// MI455X (gfx1250) — matrix operand order repaired and checked by reading, not compiled, not run on hardware
//
#include <hip/hip_runtime.h>
#include <hip/hip_bf16.h>
#include <math.h>

// ---------------------------------------------------------------------------
// Types for CDNA5 WMMA (wave32): 16x16x32 bf16 -> f32 accumulate
// ---------------------------------------------------------------------------
typedef __attribute__((ext_vector_type(16))) __bf16 v16bf;
typedef __attribute__((ext_vector_type(8)))  __bf16 v8bf;
typedef __attribute__((ext_vector_type(8)))  float  v8f;

#define ACT_NONE 0
#define ACT_GELU 1
#define ACT_SIG  2

static constexpr int NB = 64;     // N sequences
static constexpr int L  = 256;
static constexpr int DM = 256;
static constexpr int DP = 256;
static constexpr int H  = 8;
static constexpr int HD = 32;
static constexpr int NL = NB * L;     // 16384 msa tokens
static constexpr int LL = L * L;      // 65536 pair tokens

__device__ __forceinline__ float act_apply(float x, int act) {
  if (act == ACT_GELU) return 0.5f * x * (1.0f + erff(x * 0.70710678118654752f));
  if (act == ACT_SIG)  return 1.0f / (1.0f + __expf(-x));
  return x;
}

// CDNA5 async copy global->LDS (tracked by ASYNCcnt), per CDNA5 ISA.
__device__ __forceinline__ void async_b128(unsigned lds_off, unsigned long long ga) {
  asm volatile("global_load_async_to_lds_b128 %0, %1, off"
               :: "v"(lds_off), "v"(ga) : "memory");
}
__device__ __forceinline__ void wait_async0() {
  asm volatile("s_wait_asynccnt 0x0" ::: "memory");
}

__device__ __forceinline__ v16bf frag_from(const __bf16* p0, const __bf16* p1) {
  v8bf lo = *(const v8bf*)p0;
  v8bf hi = *(const v8bf*)p1;
  return __builtin_shufflevector(lo, hi, 0,1,2,3,4,5,6,7,8,9,10,11,12,13,14,15);
}

// ---------------------------------------------------------------------------
// TILED GEMM: C[M,N] = act(alpha * A[M,K] x Bt[N,K] + bias[N])
//   Block tile 128x128, K-step 32, 8 waves; wave tile 64x32 (4x2 WMMA frags).
//   A, Bt staged in double-buffered LDS via global_load_async_to_lds_b128.
//   Requires M%128==0, N%128==0, K%32==0 (true for all large GEMMs here).
// Fragment layouts per CDNA5 ISA 7.12.2 (16-bit A 16x32, B 32x16, C f32 16x16).
// ---------------------------------------------------------------------------
#define TBM 128
#define TBK 32
#define TPAD 40   // padded LDS row (elements) -> conflict-free b128 reads

__global__ __launch_bounds__(256)
void wmma_gemm_tiled(const __bf16* __restrict__ A,
                     const __bf16* __restrict__ B,
                     void* __restrict__ Cp,
                     const float* __restrict__ bias,
                     int M, int N, int K,
                     int lda, int ldb, int ldc,
                     long long sA, long long sB, long long sC,
                     float alpha, int act, int out_bf16)
{
  __shared__ __align__(16) __bf16 shA[2][TBM * TPAD];
  __shared__ __align__(16) __bf16 shB[2][TBM * TPAD];

  const int lane = threadIdx.x;         // 0..31
  const int half = lane >> 4;
  const int l15  = lane & 15;
  const int w    = threadIdx.y;         // wave 0..7
  const int wm   = w >> 2;              // 0..1  (64-row strip)
  const int wn   = w & 3;               // 0..3  (32-col strip)
  const int tid  = w * 32 + lane;       // 0..255

  const int bm0 = blockIdx.x * TBM;
  const int bn0 = blockIdx.y * TBM;
  const int z   = blockIdx.z;
  A += (long long)z * sA;
  B += (long long)z * sB;
  const long long coff = (long long)z * sC;

  const int koffA = half * 8;
  const int koffB = half * 8;
  const int nk = K / TBK;

  // stage one 128x32 A tile + 128x32 B tile into LDS buffer `buf`
  auto stage = [&](int buf, int k0) {
    #pragma unroll
    for (int s = 0; s < 2; ++s) {
      const int c   = tid + s * 256;   // 512 chunks of 16B per tile
      const int row = c >> 2;
      const int q   = (c & 3) * 8;     // element offset within K-strip
      async_b128((unsigned)(unsigned long long)(const void*)&shA[buf][row * TPAD + q],
                 (unsigned long long)(const void*)(A + (size_t)(bm0 + row) * lda + k0 + q));
      async_b128((unsigned)(unsigned long long)(const void*)&shB[buf][row * TPAD + q],
                 (unsigned long long)(const void*)(B + (size_t)(bn0 + row) * ldb + k0 + q));
    }
  };

  v8f acc[4][2];
  #pragma unroll
  for (int mf = 0; mf < 4; ++mf)
    #pragma unroll
    for (int nf = 0; nf < 2; ++nf)
      acc[mf][nf] = (v8f){0.f,0.f,0.f,0.f,0.f,0.f,0.f,0.f};

  stage(0, 0);
  for (int kt = 0; kt < nk; ++kt) {
    wait_async0();        // my stage-kt async copies have landed in LDS
    __syncthreads();      // everyone's have
    if (kt + 1 < nk) stage((kt + 1) & 1, (kt + 1) * TBK);

    const __bf16* sA_ = shA[kt & 1];
    const __bf16* sB_ = shB[kt & 1];

    v16bf bfrag[2];
    #pragma unroll
    for (int nf = 0; nf < 2; ++nf) {
      const __bf16* bp = &sB_[(wn * 32 + nf * 16 + l15) * TPAD + koffB];
      bfrag[nf] = frag_from(bp, bp + 16);    // K = {0..7, 16..23} (+8 per half)
    }
    #pragma unroll
    for (int mf = 0; mf < 4; ++mf) {
      const __bf16* ap = &sA_[(wm * 64 + mf * 16 + l15) * TPAD + koffA];
      v16bf af = frag_from(ap, ap + 16);     // K = {0..7, 16..23} (+8 per half)
      #pragma unroll
      for (int nf = 0; nf < 2; ++nf)
        acc[mf][nf] = __builtin_amdgcn_wmma_f32_16x16x32_bf16(
            false, af, false, bfrag[nf], (short)0, acc[mf][nf], false, false);
    }
    __syncthreads();      // done reading buf before it is refilled next iter
  }

  #pragma unroll
  for (int nf = 0; nf < 2; ++nf) {
    const int gcol = bn0 + wn * 32 + nf * 16 + l15;
    const float bv = bias ? bias[gcol] : 0.f;
    #pragma unroll
    for (int mf = 0; mf < 4; ++mf)
      #pragma unroll
      for (int r = 0; r < 8; ++r) {
        float x = act_apply(acc[mf][nf][r] * alpha + bv, act);
        long long ci = coff +
            (long long)(bm0 + wm * 64 + mf * 16 + r + half * 8) * ldc + gcol;
        if (out_bf16) ((__bf16*)Cp)[ci] = (__bf16)x;
        else          ((float*)Cp)[ci]  = x;
      }
  }
}

// ---------------------------------------------------------------------------
// DIRECT GEMM (1 wave = one 16x16 tile) for small / strided cases:
// scores (K=32), P*V (strided B), pair bias (N=8). Two-level batch strides.
// ---------------------------------------------------------------------------
template<bool CONTIG_B>
__global__ __launch_bounds__(128)
void wmma_gemm_direct(const __bf16* __restrict__ A,
                      const __bf16* __restrict__ B,
                      void* __restrict__ Cp,
                      const float* __restrict__ bias,
                      int M, int N, int K,
                      int lda, int ldb, int ldc,
                      long long sAo, long long sAi,
                      long long sBo, long long sBi,
                      long long sCo, long long sCi,
                      int inner, float alpha, int act, int out_bf16)
{
  const int lane = threadIdx.x;
  const int half = lane >> 4;
  const int l15  = lane & 15;

  const int tileM = blockIdx.x * 16;
  const int tileN = (blockIdx.y * blockDim.y + threadIdx.y) * 16;
  if (tileM >= M || tileN >= N) return;   // uniform per wave

  const int z  = blockIdx.z;
  const int zo = z / inner;
  const int zi = z - zo * inner;
  A += (long long)zo * sAo + (long long)zi * sAi;
  B += (long long)zo * sBo + (long long)zi * sBi;
  const long long coff = (long long)zo * sCo + (long long)zi * sCi;

  const int arow  = tileM + l15;
  const int bn    = min(tileN + l15, N - 1);   // clamp for ragged N (N=8)
  const int koffA = half * 16;
  const int koffB = half * 16;

  v8f acc = {0.f,0.f,0.f,0.f,0.f,0.f,0.f,0.f};

  for (int k0 = 0; k0 < K; k0 += 32) {
    const __bf16* ap = A + (size_t)arow * lda + (k0 + koffA);
    v16bf af = frag_from(ap, ap + 8);

    v16bf bfr;
    if (CONTIG_B) {
      const __bf16* bp = B + (size_t)bn * ldb + (k0 + koffB);
      bfr = frag_from(bp, bp + 8);
    } else {
      #pragma unroll
      for (int j = 0; j < 16; ++j)
        bfr[j] = B[(size_t)(k0 + koffB + j) * ldb + bn];
    }
    acc = __builtin_amdgcn_wmma_f32_16x16x32_bf16(
        false, af, false, bfr, (short)0, acc, false, false);
  }

  const int gcol = tileN + l15;
  const bool inb = (gcol < N);
  float bv = (bias && inb) ? bias[gcol] : 0.f;
  #pragma unroll
  for (int r = 0; r < 8; ++r) {
    if (inb) {
      float x = act_apply(acc[r] * alpha + bv, act);
      long long ci = coff + (long long)(tileM + r + half * 8) * ldc + gcol;
      if (out_bf16) ((__bf16*)Cp)[ci] = (__bf16)x;
      else          ((float*)Cp)[ci]  = x;
    }
  }
}

// ---------------------------------------------------------------------------
// Elementwise / reduction kernels
// ---------------------------------------------------------------------------
__global__ void k_conv_bf16(const float* __restrict__ X, __bf16* __restrict__ Y,
                            long long n) {
  long long i = (long long)blockIdx.x * blockDim.x + threadIdx.x;
  if (i < n) Y[i] = (__bf16)X[i];
}

// W[K,N] f32 (y = x @ W) -> Wt[N,K] bf16 (row-major, contiguous K)
__global__ void k_conv_wt(const float* __restrict__ W, __bf16* __restrict__ Wt,
                          int K, int N) {
  long long i = (long long)blockIdx.x * blockDim.x + threadIdx.x;
  if (i >= (long long)K * N) return;
  int k = (int)(i / N), n = (int)(i % N);
  Wt[(long long)n * K + k] = (__bf16)W[i];
}

// Row softmax over L=256; optional pair bias [LL,H] indexed [i*L+j, h].
__global__ void k_softmax(const float* __restrict__ S,
                          const float* __restrict__ pb,
                          __bf16* __restrict__ P) {
  const int r   = blockIdx.x;
  const int tid = threadIdx.x;          // = j
  const int i   = r % L;
  const int h   = (r / L) % H;
  __shared__ float sh[256];

  const long long base = (long long)r * L;
  float x = S[base + tid];
  if (pb) x += pb[((long long)i * L + tid) * H + h];

  sh[tid] = x; __syncthreads();
  for (int s = 128; s > 0; s >>= 1) {
    if (tid < s) sh[tid] = fmaxf(sh[tid], sh[tid + s]);
    __syncthreads();
  }
  const float m = sh[0]; __syncthreads();

  const float e = __expf(x - m);
  sh[tid] = e; __syncthreads();
  for (int s = 128; s > 0; s >>= 1) {
    if (tid < s) sh[tid] += sh[tid + s];
    __syncthreads();
  }
  P[base + tid] = (__bf16)(e / sh[0]);
}

// Residual + LayerNorm (D=256): x = resid + (gate ? gate*t : t)
__global__ void k_ln(const float* __restrict__ resid,
                     const float* __restrict__ t,
                     const float* __restrict__ gate,
                     const float* __restrict__ g,
                     const float* __restrict__ b,
                     float* __restrict__ outf,
                     __bf16* __restrict__ outb) {
  const int row = blockIdx.x;
  const int tid = threadIdx.x;
  const long long idx = (long long)row * 256 + tid;
  __shared__ float sh[256];

  float x = resid[idx] + (gate ? gate[idx] * t[idx] : t[idx]);

  sh[tid] = x; __syncthreads();
  for (int s = 128; s > 0; s >>= 1) {
    if (tid < s) sh[tid] += sh[tid + s];
    __syncthreads();
  }
  const float mean = sh[0] * (1.0f / 256.0f); __syncthreads();

  const float d = x - mean;
  sh[tid] = d * d; __syncthreads();
  for (int s = 128; s > 0; s >>= 1) {
    if (tid < s) sh[tid] += sh[tid + s];
    __syncthreads();
  }
  const float var = sh[0] * (1.0f / 256.0f);

  const float y = d * rsqrtf(var + 1e-5f) * g[tid] + b[tid];
  outf[idx] = y;
  if (outb) outb[idx] = (__bf16)y;
}

// ---------------------------------------------------------------------------
// Host-side launch helpers
// ---------------------------------------------------------------------------
static inline void gemmD(hipStream_t s, const __bf16* A, const __bf16* B, void* C,
                         const float* bias, int M, int N, int K,
                         int lda, int ldb, int ldc,
                         long long sAo, long long sAi,
                         long long sBo, long long sBi,
                         long long sCo, long long sCi,
                         int inner, int batch,
                         float alpha, int act, int outbf, bool contigB) {
  dim3 blk(32, 4, 1);
  int tilesN = (N + 15) / 16;
  dim3 grd(M / 16, (tilesN + 3) / 4, batch);
  if (contigB)
    wmma_gemm_direct<true><<<grd, blk, 0, s>>>(A, B, C, bias, M, N, K, lda, ldb, ldc,
                                               sAo, sAi, sBo, sBi, sCo, sCi,
                                               inner, alpha, act, outbf);
  else
    wmma_gemm_direct<false><<<grd, blk, 0, s>>>(A, B, C, bias, M, N, K, lda, ldb, ldc,
                                                sAo, sAi, sBo, sBi, sCo, sCi,
                                                inner, alpha, act, outbf);
}

static inline void gemmT(hipStream_t s, const __bf16* A, const __bf16* B, void* C,
                         const float* bias, int M, int N, int K,
                         int lda, int ldb, int ldc,
                         long long sA, long long sB, long long sC, int batch,
                         float alpha, int act, int outbf) {
  dim3 blk(32, 8, 1);
  dim3 grd(M / TBM, N / TBM, batch);
  wmma_gemm_tiled<<<grd, blk, 0, s>>>(A, B, C, bias, M, N, K, lda, ldb, ldc,
                                      sA, sB, sC, alpha, act, outbf);
}

extern "C" void kernel_launch(void* const* d_in, const int* in_sizes, int n_in,
                              void* d_out, int out_size, void* d_ws, size_t ws_size,
                              hipStream_t stream) {
  (void)in_sizes; (void)n_in; (void)out_size; (void)ws_size;

  int ii = 0;
  const float* msa_in  = (const float*)d_in[ii++];  // [NL, DM]
  const float* pair_in = (const float*)d_in[ii++];  // [LL, DP]
  const float* qw   = (const float*)d_in[ii++]; const float* qb   = (const float*)d_in[ii++];
  const float* kw   = (const float*)d_in[ii++]; const float* kb   = (const float*)d_in[ii++];
  const float* vw   = (const float*)d_in[ii++]; const float* vb   = (const float*)d_in[ii++];
  const float* pbw  = (const float*)d_in[ii++]; const float* pbb  = (const float*)d_in[ii++];
  const float* ow   = (const float*)d_in[ii++]; const float* ob   = (const float*)d_in[ii++];
  const float* rn_g = (const float*)d_in[ii++]; const float* rn_b = (const float*)d_in[ii++];
  const float* ca_inw = (const float*)d_in[ii++]; const float* ca_inb = (const float*)d_in[ii++];
  const float* ca_ow  = (const float*)d_in[ii++]; const float* ca_ob  = (const float*)d_in[ii++];
  const float* cn_g = (const float*)d_in[ii++]; const float* cn_b = (const float*)d_in[ii++];
  const float* ta_w = (const float*)d_in[ii++]; const float* ta_b = (const float*)d_in[ii++];
  const float* tb_w = (const float*)d_in[ii++]; const float* tb_b = (const float*)d_in[ii++];
  const float* tg_w = (const float*)d_in[ii++]; const float* tg_b = (const float*)d_in[ii++];
  const float* tn_g = (const float*)d_in[ii++]; const float* tn_b = (const float*)d_in[ii++];
  const float* mf1_w = (const float*)d_in[ii++]; const float* mf1_b = (const float*)d_in[ii++];
  const float* mf2_w = (const float*)d_in[ii++]; const float* mf2_b = (const float*)d_in[ii++];
  const float* pf1_w = (const float*)d_in[ii++]; const float* pf1_b = (const float*)d_in[ii++];
  const float* pf2_w = (const float*)d_in[ii++]; const float* pf2_b = (const float*)d_in[ii++];
  const float* mn_g = (const float*)d_in[ii++]; const float* mn_b = (const float*)d_in[ii++];
  const float* pn_g = (const float*)d_in[ii++]; const float* pn_b = (const float*)d_in[ii++];

  float* msa_out  = (float*)d_out;                       // [NL, DM]
  float* pair_out = (float*)d_out + (long long)NL * DM;  // [LL, DP]

  char* wsb = (char*)d_ws;
  size_t off = 0;
  auto alloc = [&](size_t bytes) -> void* {
    void* p = wsb + off;
    off = (off + bytes + 255) & ~(size_t)255;
    return p;
  };

  // weights (bf16, transposed to [out,in])
  __bf16* qw_t   = (__bf16*)alloc((size_t)DM * DM * 2);
  __bf16* kw_t   = (__bf16*)alloc((size_t)DM * DM * 2);
  __bf16* vw_t   = (__bf16*)alloc((size_t)DM * DM * 2);
  __bf16* ow_t   = (__bf16*)alloc((size_t)DM * DM * 2);
  __bf16* pbw_t  = (__bf16*)alloc((size_t)DP * H * 2);
  __bf16* cain_t = (__bf16*)alloc((size_t)DM * 3 * DM * 2);
  __bf16* caow_t = (__bf16*)alloc((size_t)DM * DM * 2);
  __bf16* ta_t   = (__bf16*)alloc((size_t)DP * DP * 2);
  __bf16* tb_t   = (__bf16*)alloc((size_t)DP * DP * 2);
  __bf16* tg_t   = (__bf16*)alloc((size_t)DP * DP * 2);
  __bf16* mf1_t  = (__bf16*)alloc((size_t)DM * 4 * DM * 2);
  __bf16* mf2_t  = (__bf16*)alloc((size_t)4 * DM * DM * 2);
  __bf16* pf1_t  = (__bf16*)alloc((size_t)DP * 4 * DP * 2);
  __bf16* pf2_t  = (__bf16*)alloc((size_t)4 * DP * DP * 2);

  // activations
  __bf16* msa_bf  = (__bf16*)alloc((size_t)NL * DM * 2);
  __bf16* pair_bf = (__bf16*)alloc((size_t)LL * DP * 2);
  __bf16* qkv_bf  = (__bf16*)alloc((size_t)NL * 3 * DM * 2);
  float*  scores  = (float*)alloc((size_t)NB * H * L * L * 4); // also pair-FFN hidden bf16
  __bf16* probs   = (__bf16*)alloc((size_t)NB * H * L * L * 2);// also upd f32
  float*  pbias   = (float*)alloc((size_t)LL * H * 4);
  __bf16* ao_bf   = (__bf16*)alloc((size_t)NL * DM * 2);
  float*  tmpf    = (float*)alloc((size_t)LL * DP * 4);
  __bf16* mh_bf   = (__bf16*)alloc((size_t)NL * 4 * DM * 2);
  __bf16* left_bf = (__bf16*)alloc((size_t)LL * DP * 2);
  __bf16* right_bf= (__bf16*)alloc((size_t)LL * DP * 2);
  float*  gatef   = (float*)alloc((size_t)LL * DP * 4);
  float*  pair2   = (float*)alloc((size_t)LL * DP * 4);
  float*  msaA    = (float*)alloc((size_t)NL * DM * 4);
  float*  msaB    = (float*)alloc((size_t)NL * DM * 4);

  __bf16* ph_bf = (__bf16*)scores;   // [LL, 1024] bf16 reuses scores bytes
  float*  updf  = (float*)probs;     // [LL, L] f32 reuses probs bytes

  auto conv = [&](const float* x, __bf16* y, long long n) {
    k_conv_bf16<<<dim3((unsigned)((n + 255) / 256)), dim3(256), 0, stream>>>(x, y, n);
  };
  auto convT = [&](const float* w, __bf16* wt, int K, int N) {
    long long n = (long long)K * N;
    k_conv_wt<<<dim3((unsigned)((n + 255) / 256)), dim3(256), 0, stream>>>(w, wt, K, N);
  };

  // ----- precision prep -----
  conv(msa_in, msa_bf, (long long)NL * DM);
  conv(pair_in, pair_bf, (long long)LL * DP);
  convT(qw, qw_t, DM, DM);     convT(kw, kw_t, DM, DM);
  convT(vw, vw_t, DM, DM);     convT(ow, ow_t, DM, DM);
  convT(pbw, pbw_t, DP, H);
  convT(ca_inw, cain_t, DM, 3 * DM);
  convT(ca_ow, caow_t, DM, DM);
  convT(ta_w, ta_t, DP, DP);   convT(tb_w, tb_t, DP, DP);
  convT(tg_w, tg_t, DP, DP);
  convT(mf1_w, mf1_t, DM, 4 * DM);
  convT(mf2_w, mf2_t, 4 * DM, DM);
  convT(pf1_w, pf1_t, DP, 4 * DP);
  convT(pf2_w, pf2_t, 4 * DP, DP);

  const float iscale = 0.17677669529663689f;  // 1/sqrt(HD)
  __bf16* q_bf = qkv_bf;
  __bf16* k_bf = qkv_bf + (size_t)NL * DM;
  __bf16* v_bf = qkv_bf + (size_t)2 * NL * DM;

  // ===== MSA row attention with pair bias =====
  gemmT(stream, msa_bf, qw_t, q_bf, qb, NL, DM, DM, DM, DM, DM, 0,0,0,1, 1.f, ACT_NONE, 1);
  gemmT(stream, msa_bf, kw_t, k_bf, kb, NL, DM, DM, DM, DM, DM, 0,0,0,1, 1.f, ACT_NONE, 1);
  gemmT(stream, msa_bf, vw_t, v_bf, vb, NL, DM, DM, DM, DM, DM, 0,0,0,1, 1.f, ACT_NONE, 1);
  gemmD(stream, pair_bf, pbw_t, pbias, pbb, LL, H, DP, DP, DP, H,
        0,0,0,0,0,0, 1, 1, 1.f, ACT_NONE, 0, true);

  gemmD(stream, q_bf, k_bf, scores, nullptr, L, L, HD, DM, DM, L,
        (long long)L * DM, HD, (long long)L * DM, HD,
        (long long)H * L * L, (long long)L * L,
        H, NB * H, iscale, ACT_NONE, 0, true);
  k_softmax<<<dim3(NB * H * L), dim3(256), 0, stream>>>(scores, pbias, probs);
  gemmD(stream, probs, v_bf, ao_bf, nullptr, L, HD, L, L, DM, DM,
        (long long)H * L * L, (long long)L * L, (long long)L * DM, HD,
        (long long)L * DM, HD,
        H, NB * H, 1.f, ACT_NONE, 1, false);
  gemmT(stream, ao_bf, ow_t, tmpf, ob, NL, DM, DM, DM, DM, DM, 0,0,0,1, 1.f, ACT_NONE, 0);
  k_ln<<<dim3(NL), dim3(256), 0, stream>>>(msa_in, tmpf, nullptr, rn_g, rn_b, msaA, msa_bf);

  // ===== column attention (fused in-proj) =====
  gemmT(stream, msa_bf, cain_t, qkv_bf, ca_inb, NL, 3 * DM, DM, DM, DM, 3 * DM,
        0,0,0,1, 1.f, ACT_NONE, 1);
  gemmD(stream, qkv_bf, qkv_bf + DM, scores, nullptr, L, L, HD, 3 * DM, 3 * DM, L,
        (long long)L * 3 * DM, HD, (long long)L * 3 * DM, HD,
        (long long)H * L * L, (long long)L * L,
        H, NB * H, iscale, ACT_NONE, 0, true);
  k_softmax<<<dim3(NB * H * L), dim3(256), 0, stream>>>(scores, nullptr, probs);
  gemmD(stream, probs, qkv_bf + 2 * DM, ao_bf, nullptr, L, HD, L, L, 3 * DM, DM,
        (long long)H * L * L, (long long)L * L, (long long)L * 3 * DM, HD,
        (long long)L * DM, HD,
        H, NB * H, 1.f, ACT_NONE, 1, false);
  gemmT(stream, ao_bf, caow_t, tmpf, ca_ob, NL, DM, DM, DM, DM, DM, 0,0,0,1,
        1.f, ACT_NONE, 0);
  k_ln<<<dim3(NL), dim3(256), 0, stream>>>(msaA, tmpf, nullptr, cn_g, cn_b, msaB, msa_bf);

  // ===== MSA FFN =====
  gemmT(stream, msa_bf, mf1_t, mh_bf, mf1_b, NL, 4 * DM, DM, DM, DM, 4 * DM,
        0,0,0,1, 1.f, ACT_GELU, 1);
  gemmT(stream, mh_bf, mf2_t, tmpf, mf2_b, NL, DM, 4 * DM, 4 * DM, 4 * DM, DM,
        0,0,0,1, 1.f, ACT_NONE, 0);
  k_ln<<<dim3(NL), dim3(256), 0, stream>>>(msaB, tmpf, nullptr, mn_g, mn_b,
                                           msa_out, nullptr);

  // ===== triangle multiplicative update =====
  gemmT(stream, pair_bf, ta_t, left_bf, ta_b, LL, DP, DP, DP, DP, DP, 0,0,0,1,
        1.f, ACT_GELU, 1);
  gemmT(stream, pair_bf, tb_t, right_bf, tb_b, LL, DP, DP, DP, DP, DP, 0,0,0,1,
        1.f, ACT_GELU, 1);
  gemmT(stream, pair_bf, tg_t, gatef, tg_b, LL, DP, DP, DP, DP, DP, 0,0,0,1,
        1.f, ACT_SIG, 0);
  // upd[l] = left_l x right_l^T  (batched over l via tiled kernel)
  gemmT(stream, left_bf, right_bf, updf, nullptr, L, L, DP, DP, DP, L,
        (long long)L * DP, (long long)L * DP, (long long)L * L, L,
        1.f, ACT_NONE, 0);
  k_ln<<<dim3(LL), dim3(256), 0, stream>>>(pair_in, updf, gatef, tn_g, tn_b,
                                           pair2, pair_bf);

  // ===== pair FFN =====
  gemmT(stream, pair_bf, pf1_t, ph_bf, pf1_b, LL, 4 * DP, DP, DP, DP, 4 * DP,
        0,0,0,1, 1.f, ACT_GELU, 1);
  gemmT(stream, ph_bf, pf2_t, tmpf, pf2_b, LL, DP, 4 * DP, 4 * DP, 4 * DP, DP,
        0,0,0,1, 1.f, ACT_NONE, 0);
  k_ln<<<dim3(LL), dim3(256), 0, stream>>>(pair2, tmpf, nullptr, pn_g, pn_b,
                                           pair_out, nullptr);
}
